// TopKMHAIntervention_9028021256289
// MI455X (gfx1250) — matrix operand order repaired and checked by reading, not compiled, not run on hardware
//
#include <hip/hip_runtime.h>

// ---------------------------------------------------------------------------
// Top-K MHA for MI455X (gfx1250, wave32, WMMA).
// GEMMs use v_wmma_f32_16x16x32_bf16 (fp32 accumulate), 32x64 tile per wave
// (8 WMMA per k-step on 12 b128 loads). Attention A*V is done sparsely
// (k=16 gathered rows of fp32 V). The kernel is bound by the 403MB dense
// topk_attn store plus ~52 GFLOP of bf16 GEMM.
// ---------------------------------------------------------------------------

#define Bq   8
#define Nseq 1024
#define Cdim 768
#define Hh   12
#define HD   64
#define TOPK 16

typedef unsigned short u16;
typedef __attribute__((ext_vector_type(16))) __bf16 v16bf;
typedef __attribute__((ext_vector_type(8)))  float  v8f;
typedef __attribute__((ext_vector_type(8)))  u16    us8;
typedef __attribute__((ext_vector_type(16))) u16    us16;

__device__ __forceinline__ u16 f2bf(float f) {
    unsigned u = __float_as_uint(f);
    u += 0x7FFFu + ((u >> 16) & 1u);   // round-to-nearest-even
    return (u16)(u >> 16);
}

// A-matrix 16x32 bf16 fragment: lanes 0-15 hold row M=lane,
// elems e = K[k0+e]; lanes 16-31 are the same rows with K[k0+16+e]
// (matches load_frag_b). Two contiguous 16B loads per lane.
__device__ __forceinline__ v16bf load_frag_a(const u16* __restrict__ base,
                                             int ld, int row, int k0, int lane) {
    int r  = row + (lane & 15);
    int ko = k0 + ((lane >> 4) << 4);          // +16 for upper half-wave
    const u16* p = base + (size_t)r * ld + ko;
    union { us8 h[2]; v16bf v; } u;
    u.h[0] = *(const us8*)(p);
    u.h[1] = *(const us8*)(p + 8);
    return u.v;
}

// B-matrix 32x16 bf16 fragment with B[k][n] = W[n][k] (K-contiguous rows):
// lanes 0-15: elems e = K[k0+e], N = ncol+lane; lanes 16-31: K[k0+16+e].
// One contiguous 32B load per lane.
__device__ __forceinline__ v16bf load_frag_b(const u16* __restrict__ base,
                                             int ld, int ncol, int k0, int lane) {
    int n  = ncol + (lane & 15);
    int ko = k0 + ((lane >> 4) << 4);          // +16 for upper half-wave
    const u16* p = base + (size_t)n * ld + ko;
    union { us16 w; v16bf v; } u;
    u.w = *(const us16*)(p);
    return u.v;
}

__device__ __forceinline__ v8f wmma_bf16(v16bf a, v16bf b, v8f c) {
    return __builtin_amdgcn_wmma_f32_16x16x32_bf16(false, a, false, b,
                                                   (short)0, c, false, false);
}

// ---------------------------------------------------------------------------
__global__ void cvt_bf16(const float* __restrict__ in, u16* __restrict__ out, int n) {
    int i = blockIdx.x * blockDim.x + threadIdx.x;
    if (i < n) out[i] = f2bf(in[i]);
}

// ---------------------------------------------------------------------------
// QKV projection: out[m][n3] = sum_k X[m][k] * W[n3][k] + bias[n3]
// One wave computes a 32x64 tile (2 A-frags x 4 B-frags = 8 WMMA per k-step).
// Writes Q,K as bf16 [B,H,N,hd], V as f32 [B,H,N,hd].
__global__ void qkv_gemm(const u16* __restrict__ X, const u16* __restrict__ W,
                         const float* __restrict__ bias,
                         u16* __restrict__ Qo, u16* __restrict__ Ko,
                         float* __restrict__ Vo) {
    const int lane  = threadIdx.x;
    const int mtile = blockIdx.x * 32;
    const int nbase = blockIdx.y * 64;
    v8f acc[2][4] = {};
    for (int k0 = 0; k0 < Cdim; k0 += 32) {
        v16bf a0 = load_frag_a(X, Cdim, mtile,      k0, lane);
        v16bf a1 = load_frag_a(X, Cdim, mtile + 16, k0, lane);
#pragma unroll
        for (int s = 0; s < 4; ++s) {
            v16bf b = load_frag_b(W, Cdim, nbase + s * 16, k0, lane);
            acc[0][s] = wmma_bf16(a0, b, acc[0][s]);
            acc[1][s] = wmma_bf16(a1, b, acc[1][s]);
        }
    }
    // C/D layout: VGPR r, lanes 0-15 -> (M=r, N=lane); lanes 16-31 -> (M=8+r).
    const int ln   = lane & 15;
    const int moff = (lane >> 4) << 3;
#pragma unroll
    for (int hm = 0; hm < 2; ++hm) {
#pragma unroll
        for (int s = 0; s < 4; ++s) {
            int n3    = nbase + s * 16 + ln;
            int which = n3 / Cdim;             // uniform within a 16-wide tile
            int cc    = n3 - which * Cdim;
            int h = cc >> 6, d = cc & 63;
            float bv = bias[n3];
#pragma unroll
            for (int r = 0; r < 8; ++r) {
                int m  = mtile + hm * 16 + moff + r;
                int b_ = m >> 10, nq = m & 1023;
                size_t dst = (((size_t)b_ * Hh + h) * Nseq + nq) * HD + d;
                float v = acc[hm][s][r] + bv;
                if (which == 0)      Qo[dst] = f2bf(v);
                else if (which == 1) Ko[dst] = f2bf(v);
                else                 Vo[dst] = v;
            }
        }
    }
}

// ---------------------------------------------------------------------------
// Per (b,h,16-query-row block): scores via WMMA into LDS, softmax, top-16
// (lower-index tie-break, matching jax.lax.top_k), renormalize, stream the
// dense topk row to global, sparse A*V into bf16 attn-out [B,N,C].
__global__ void attn_topk(const u16* __restrict__ Qb_, const u16* __restrict__ Kb_,
                          const float* __restrict__ Vb_,
                          float* __restrict__ topk_out, u16* __restrict__ AOut) {
    extern __shared__ char smem_raw[];
    float* S  = (float*)smem_raw;        // [16][1024] scores/probs (64 KB)
    float* TV = S + 16 * 1024;           // [16][16] top values
    int*   TI = (int*)(TV + 256);        // [16][16] top indices

    const int lane  = threadIdx.x & 31;
    const int wave  = threadIdx.x >> 5;
    const int bid   = blockIdx.x;
    const int qt    = bid & 63;          // 64 query tiles per (b,h)
    const int bh    = bid >> 6;          // b*H + h
    const int qrow0 = qt * 16;
    const u16*   Qp = Qb_ + (size_t)bh * Nseq * HD;
    const u16*   Kp = Kb_ + (size_t)bh * Nseq * HD;
    const float* Vp = Vb_ + (size_t)bh * Nseq * HD;
    const float scale = 0.125f;          // 1/sqrt(64)

    // Phase 1: S = scale * Q K^T  (64 column tiles; 4 waves -> 16 tiles each)
    for (int t = wave; t < 64; t += 4) {
        v8f c = {};
#pragma unroll
        for (int kk = 0; kk < 2; ++kk) {
            v16bf a = load_frag_a(Qp, HD, qrow0, kk * 32, lane);   // t-invariant
            v16bf b = load_frag_b(Kp, HD, t * 16, kk * 32, lane);
            c = wmma_bf16(a, b, c);
        }
        int ncol = t * 16 + (lane & 15);
        int mo   = (lane >> 4) << 3;
#pragma unroll
        for (int r = 0; r < 8; ++r)
            S[(mo + r) * 1024 + ncol] = c[r] * scale;
    }
    __syncthreads();

    // One wave per row, 4 rows per wave.
    for (int r = wave; r < 16; r += 4) {
        float* Sr = S + r * 1024;

        // softmax over 1024
        float mx = -3.4e38f;
        for (int j = lane; j < 1024; j += 32) mx = fmaxf(mx, Sr[j]);
#pragma unroll
        for (int o = 16; o > 0; o >>= 1) mx = fmaxf(mx, __shfl_xor(mx, o, 32));
        float sum = 0.f;
        for (int j = lane; j < 1024; j += 32) {
            float e = __expf(Sr[j] - mx); Sr[j] = e; sum += e;
        }
#pragma unroll
        for (int o = 16; o > 0; o >>= 1) sum += __shfl_xor(sum, o, 32);
        float inv = 1.0f / sum;
        for (int j = lane; j < 1024; j += 32) Sr[j] *= inv;

        // iterative top-16 (prefer larger value, then lower index)
        float den = 0.f;
#pragma unroll 1
        for (int t = 0; t < TOPK; ++t) {
            float best = -2.f; int bidx = 1 << 30;
            for (int j = lane; j < 1024; j += 32) {
                float v = Sr[j];
                if (v > best) { best = v; bidx = j; }
            }
#pragma unroll
            for (int o = 16; o > 0; o >>= 1) {
                float ov = __shfl_xor(best, o, 32);
                int   oi = __shfl_xor(bidx, o, 32);
                if (ov > best || (ov == best && oi < bidx)) { best = ov; bidx = oi; }
            }
            if (lane == 0) { TV[r * 16 + t] = best; TI[r * 16 + t] = bidx; Sr[bidx] = -1.f; }
            den += best;
        }
        den = fmaxf(den, 1e-6f);

        // rebuild dense renormalized row in LDS, stream to global (b128)
        for (int j = lane; j < 1024; j += 32) Sr[j] = 0.f;
        if (lane < TOPK) Sr[TI[r * 16 + lane]] = TV[r * 16 + lane] / den;
        float* orow = topk_out + ((size_t)bh * Nseq + qrow0 + r) * Nseq;
        for (int j = lane * 4; j < 1024; j += 128)
            *(float4*)(orow + j) = *(const float4*)(Sr + j);

        // sparse A*V: 16 gathered V rows, fp32; lane covers d and d+32
        float a0 = 0.f, a1 = 0.f;
#pragma unroll 1
        for (int t = 0; t < TOPK; ++t) {
            float w = TV[r * 16 + t] / den;
            const float* vr = Vp + (size_t)TI[r * 16 + t] * HD;
            a0 += w * vr[lane];
            a1 += w * vr[lane + 32];
        }
        int b_ = bh / Hh, h = bh % Hh;
        size_t mrow = (size_t)b_ * Nseq + qrow0 + r;
        u16* arow = AOut + mrow * Cdim + h * HD;
        arow[lane]      = f2bf(a0);
        arow[lane + 32] = f2bf(a1);
    }
}

// ---------------------------------------------------------------------------
// Output projection: O[m][n] = sum_k A[m][k] * W[n][k] + bias[n]  (fp32 out)
// Same 32x64-per-wave tiling as qkv_gemm.
__global__ void out_gemm(const u16* __restrict__ X, const u16* __restrict__ W,
                         const float* __restrict__ bias, float* __restrict__ O) {
    const int lane  = threadIdx.x;
    const int mtile = blockIdx.x * 32;
    const int nbase = blockIdx.y * 64;
    v8f acc[2][4] = {};
    for (int k0 = 0; k0 < Cdim; k0 += 32) {
        v16bf a0 = load_frag_a(X, Cdim, mtile,      k0, lane);
        v16bf a1 = load_frag_a(X, Cdim, mtile + 16, k0, lane);
#pragma unroll
        for (int s = 0; s < 4; ++s) {
            v16bf b = load_frag_b(W, Cdim, nbase + s * 16, k0, lane);
            acc[0][s] = wmma_bf16(a0, b, acc[0][s]);
            acc[1][s] = wmma_bf16(a1, b, acc[1][s]);
        }
    }
    const int ln   = lane & 15;
    const int moff = (lane >> 4) << 3;
#pragma unroll
    for (int hm = 0; hm < 2; ++hm) {
#pragma unroll
        for (int s = 0; s < 4; ++s) {
            int n = nbase + s * 16 + ln;
            float bv = bias[n];
#pragma unroll
            for (int r = 0; r < 8; ++r) {
                int m = mtile + hm * 16 + moff + r;
                O[(size_t)m * Cdim + n] = acc[hm][s][r] + bv;
            }
        }
    }
}

// ---------------------------------------------------------------------------
extern "C" void kernel_launch(void* const* d_in, const int* in_sizes, int n_in,
                              void* d_out, int out_size, void* d_ws, size_t ws_size,
                              hipStream_t stream) {
    (void)in_sizes; (void)n_in; (void)out_size; (void)ws_size;
    const float* query = (const float*)d_in[0];
    const float* w_in  = (const float*)d_in[1];
    const float* b_in  = (const float*)d_in[2];
    const float* w_out = (const float*)d_in[3];
    const float* b_out = (const float*)d_in[4];

    // workspace carve-out (256B aligned), total ~81 MB
    char* ws = (char*)d_ws;
    size_t off = 0;
    auto alloc = [&](size_t bytes) -> char* {
        char* p = ws + off;
        off = (off + bytes + 255) & ~(size_t)255;
        return p;
    };
    u16*   xq = (u16*)alloc((size_t)Bq * Nseq * Cdim * 2);      // query bf16
    u16*   w1 = (u16*)alloc((size_t)3 * Cdim * Cdim * 2);       // in_proj_w bf16
    u16*   w2 = (u16*)alloc((size_t)Cdim * Cdim * 2);           // out_proj_w bf16
    u16*   qb = (u16*)alloc((size_t)Bq * Hh * Nseq * HD * 2);   // Q bf16
    u16*   kb = (u16*)alloc((size_t)Bq * Hh * Nseq * HD * 2);   // K bf16
    float* vb = (float*)alloc((size_t)Bq * Hh * Nseq * HD * 4); // V f32
    u16*   ao = (u16*)alloc((size_t)Bq * Nseq * Cdim * 2);      // attn-out bf16

    float* out_o  = (float*)d_out;
    float* topk_o = out_o + (size_t)Bq * Nseq * Cdim;

    int nq = Bq * Nseq * Cdim;
    cvt_bf16<<<(nq + 255) / 256, 256, 0, stream>>>(query, xq, nq);
    int n1 = 3 * Cdim * Cdim;
    cvt_bf16<<<(n1 + 255) / 256, 256, 0, stream>>>(w_in, w1, n1);
    int n2 = Cdim * Cdim;
    cvt_bf16<<<(n2 + 255) / 256, 256, 0, stream>>>(w_out, w2, n2);

    qkv_gemm<<<dim3(Bq * Nseq / 32, 3 * Cdim / 64), 32, 0, stream>>>(
        xq, w1, b_in, qb, kb, vb);

    size_t shm = 16 * 1024 * 4 + 256 * 4 + 256 * 4;   // 66 KB dynamic LDS
    attn_topk<<<Bq * Hh * (Nseq / 16), 128, shm, stream>>>(
        qb, kb, vb, topk_o, ao);

    out_gemm<<<dim3(Bq * Nseq / 32, Cdim / 64), 32, 0, stream>>>(
        ao, w2, b_out, out_o);
}
